// GroupedLinear_12841952215331
// MI455X (gfx1250) — matrix operand order repaired and checked by reading, not compiled, not run on hardware
//
#include <hip/hip_runtime.h>

typedef __attribute__((ext_vector_type(16))) __bf16 v16bf;
typedef __attribute__((ext_vector_type(8)))  __bf16 bf16x8;
typedef __attribute__((ext_vector_type(4)))  __bf16 bf16x4;
typedef __attribute__((ext_vector_type(8)))  float  v8f;
typedef __attribute__((ext_vector_type(4)))  float  f32x4;

// Problem constants (fixed by the reference)
constexpr int E   = 8;
constexpr int IN  = 2048;
constexpr int OUT = 5632;
constexpr int T   = 16384;

// Tiling
constexpr int BM = 128;
constexpr int BN = 128;
constexpr int BK = 32;                 // one 16x16x32 bf16 WMMA K-depth
constexpr int LDS_STRIDE = BK + 8;     // 40 elems -> 80B rows: 16B-aligned, bank-staggered

union FragU { v16bf v; bf16x8 h[2]; };

__global__ __launch_bounds__(256) void grouped_gemm_bf16x3(
    const float* __restrict__ x,      // [T, IN]
    const float* __restrict__ w,      // [E, OUT, IN]
    const int*   __restrict__ offs,   // [E+1]
    float*       __restrict__ out)    // [T, OUT]
{
    __shared__ __bf16 Ah[BM * LDS_STRIDE];
    __shared__ __bf16 Al[BM * LDS_STRIDE];
    __shared__ __bf16 Bh[BN * LDS_STRIDE];
    __shared__ __bf16 Bl[BN * LDS_STRIDE];

    const int tid     = threadIdx.x;
    const int rowBase = blockIdx.y * BM;   // token tile start
    const int colBase = blockIdx.x * BN;   // output-feature tile start

    // Expert lookup (uniform per block; offsets are monotonic, groups are
    // multiples of 512 so a 128-row tile never straddles an expert boundary).
    int e = 0;
    #pragma unroll
    for (int i = 1; i < E; ++i) e += (rowBase >= offs[i]) ? 1 : 0;
    const float* __restrict__ we = w + (size_t)e * OUT * IN;

    const int wave  = tid >> 5;
    const int lane  = tid & 31;
    const int half  = lane >> 4;   // 0: lanes 0-15, 1: lanes 16-31
    const int l16   = lane & 15;
    const int wm0   = (wave & 3) * 32;   // wave M offset within block tile
    const int wn0   = (wave >> 2) * 64;  // wave N offset within block tile

    // Per-thread staging addresses (constant across K iterations)
    const int f0    = tid;                 // float4 id base, stride 256
    const int ar[4] = { (f0) >> 3, (f0 + 256) >> 3, (f0 + 512) >> 3, (f0 + 768) >> 3 };
    const int ac[4] = { ((f0) & 7) << 2, ((f0 + 256) & 7) << 2,
                        ((f0 + 512) & 7) << 2, ((f0 + 768) & 7) << 2 };

    v8f acc[2][4];
    #pragma unroll
    for (int mi = 0; mi < 2; ++mi)
        #pragma unroll
        for (int ni = 0; ni < 4; ++ni)
            acc[mi][ni] = (v8f){};

    // ---- Register prefetch pipeline: global loads for slab k0 overlap the
    //      WMMA compute of slab k0-BK.
    f32x4 aReg[4], bReg[4];
    #pragma unroll
    for (int i = 0; i < 4; ++i) {
        aReg[i] = *(const f32x4*)(x  + (size_t)(rowBase + ar[i]) * IN + ac[i]);
        bReg[i] = *(const f32x4*)(we + (size_t)(colBase + ar[i]) * IN + ac[i]);
    }

    for (int k0 = 0; k0 < IN; k0 += BK) {
        // ---- Stage current slab: fp32 -> bf16 hi/lo into LDS
        #pragma unroll
        for (int i = 0; i < 4; ++i) {
            bf16x4 hi, lo;
            #pragma unroll
            for (int j = 0; j < 4; ++j) {
                __bf16 h = (__bf16)aReg[i][j];
                hi[j] = h;
                lo[j] = (__bf16)(aReg[i][j] - (float)h);
            }
            *(bf16x4*)&Ah[ar[i] * LDS_STRIDE + ac[i]] = hi;
            *(bf16x4*)&Al[ar[i] * LDS_STRIDE + ac[i]] = lo;
            #pragma unroll
            for (int j = 0; j < 4; ++j) {
                __bf16 h = (__bf16)bReg[i][j];
                hi[j] = h;
                lo[j] = (__bf16)(bReg[i][j] - (float)h);
            }
            *(bf16x4*)&Bh[ar[i] * LDS_STRIDE + ac[i]] = hi;
            *(bf16x4*)&Bl[ar[i] * LDS_STRIDE + ac[i]] = lo;
        }
        __syncthreads();

        // ---- Issue next slab's global loads; latency hides under WMMAs below
        const int kn = k0 + BK;
        if (kn < IN) {
            #pragma unroll
            for (int i = 0; i < 4; ++i) {
                aReg[i] = *(const f32x4*)(x  + (size_t)(rowBase + ar[i]) * IN + kn + ac[i]);
                bReg[i] = *(const f32x4*)(we + (size_t)(colBase + ar[i]) * IN + kn + ac[i]);
            }
        }

        // ---- A fragments (16x32 bf16): lane<16 -> row M=lane, Ks {0-7,16-23};
        //      lane>=16 -> Ks {8-15,24-31}; two 16B LDS chunks each.
        v16bf afh[2], afl[2];
        #pragma unroll
        for (int mi = 0; mi < 2; ++mi) {
            const int row = wm0 + mi * 16 + l16;
            const __bf16* ph = &Ah[row * LDS_STRIDE + half * 8];
            const __bf16* pl = &Al[row * LDS_STRIDE + half * 8];
            FragU u;
            u.h[0] = *(const bf16x8*)(ph);
            u.h[1] = *(const bf16x8*)(ph + 16);
            afh[mi] = u.v;
            u.h[0] = *(const bf16x8*)(pl);
            u.h[1] = *(const bf16x8*)(pl + 16);
            afl[mi] = u.v;
        }
        // ---- B fragments (32x16 bf16): lane -> column N=lane&15,
        //      Ks {half*8 + 0..7, 16 + half*8 + 0..7} (matches A).
        v16bf bfh[4], bfl[4];
        #pragma unroll
        for (int ni = 0; ni < 4; ++ni) {
            const int col = wn0 + ni * 16 + l16;
            const __bf16* ph = &Bh[col * LDS_STRIDE + half * 8];
            const __bf16* pl = &Bl[col * LDS_STRIDE + half * 8];
            FragU u;
            u.h[0] = *(const bf16x8*)(ph);
            u.h[1] = *(const bf16x8*)(ph + 16);
            bfh[ni] = u.v;
            u.h[0] = *(const bf16x8*)(pl);
            u.h[1] = *(const bf16x8*)(pl + 16);
            bfl[ni] = u.v;
        }

        // ---- fp32-via-bf16 3-product WMMA: ah*bh + ah*bl + al*bh
        #pragma unroll
        for (int mi = 0; mi < 2; ++mi) {
            #pragma unroll
            for (int ni = 0; ni < 4; ++ni) {
                v8f c = acc[mi][ni];
                c = __builtin_amdgcn_wmma_f32_16x16x32_bf16(
                        false, afh[mi], false, bfh[ni], (short)0, c, false, false);
                c = __builtin_amdgcn_wmma_f32_16x16x32_bf16(
                        false, afh[mi], false, bfl[ni], (short)0, c, false, false);
                c = __builtin_amdgcn_wmma_f32_16x16x32_bf16(
                        false, afl[mi], false, bfh[ni], (short)0, c, false, false);
                acc[mi][ni] = c;
            }
        }
        __syncthreads();
    }

    // ---- Epilogue: 16x16 f32 C layout: VGPR r -> row (r + half*8), col lane&15
    #pragma unroll
    for (int mi = 0; mi < 2; ++mi) {
        #pragma unroll
        for (int ni = 0; ni < 4; ++ni) {
            const int row0 = rowBase + wm0 + mi * 16 + half * 8;
            const int col  = colBase + wn0 + ni * 16 + l16;
            #pragma unroll
            for (int r = 0; r < 8; ++r) {
                out[(size_t)(row0 + r) * OUT + col] = acc[mi][ni][r];
            }
        }
    }
}

extern "C" void kernel_launch(void* const* d_in, const int* in_sizes, int n_in,
                              void* d_out, int out_size, void* d_ws, size_t ws_size,
                              hipStream_t stream) {
    (void)in_sizes; (void)n_in; (void)out_size; (void)d_ws; (void)ws_size;
    const float* x    = (const float*)d_in[0];
    const float* w    = (const float*)d_in[1];
    const int*   offs = (const int*)d_in[2];
    float*       out  = (float*)d_out;

    dim3 grid(OUT / BN, T / BM);   // x-major: consecutive blocks reuse one expert's W in L2
    grouped_gemm_bf16x3<<<grid, 256, 0, stream>>>(x, w, offs, out);
}
